// RBNPlus_46033459478884
// MI455X (gfx1250) — matrix operand order repaired and checked by reading, not compiled, not run on hardware
//
#include <hip/hip_runtime.h>
#include <hip/hip_bf16.h>

typedef __attribute__((ext_vector_type(16))) _Float16 v16h;
typedef __attribute__((ext_vector_type(8)))  float    v8f;
typedef _Float16 h16;

#define B_   8
#define CF   4096
#define CTK  512
#define DM_  256
#define HH   8
#define DH   32
#define NSL  64

// ---------------- device helpers ----------------
__device__ inline float gelu_f(float x) {
  return 0.5f * x * (1.0f + erff(x * 0.70710678118654752f));
}
__device__ inline float sigm(float x) { return 1.f / (1.f + expf(-x)); }
__device__ inline float wsum(float v) {
  #pragma unroll
  for (int m = 16; m; m >>= 1) v += __shfl_xor(v, m, 32);
  return v;
}
__device__ inline float wmaxf(float v) {
  #pragma unroll
  for (int m = 16; m; m >>= 1) v = fmaxf(v, __shfl_xor(v, m, 32));
  return v;
}
__device__ inline int wsumi(int v) {
  #pragma unroll
  for (int m = 16; m; m >>= 1) v += __shfl_xor(v, m, 32);
  return v;
}
__device__ inline v8f wmma16(v16h a, v16h b, v8f c) {
  return __builtin_amdgcn_wmma_f32_16x16x32_f16(false, a, false, b, (short)0, c, false, false);
}
// A fragment: rows m0..m0+15 x K cols k0..k0+31, X row-major f16, stride lda.
// lanes 0-15: row m0+lane, halves K {0..7,16..23}; lanes 16-31: same rows, K {8..15,24..31}
__device__ inline v16h frag_a(const h16* X, int lda, int m0, int k0) {
  int lane = threadIdx.x & 31;
  int lh = lane & 15, hb = (lane >> 4) << 3;
  const h16* r = X + (long)(m0 + lh) * lda + k0 + hb;
  v16h f;
  #pragma unroll
  for (int i = 0; i < 8; ++i) { f[i] = r[i]; f[8 + i] = r[16 + i]; }
  return f;
}
// B fragment for D = X @ W^T : W stored row-major [N,K] stride ldw.
// lane holds column n0+(lane&15); halves K {0..7,16..23} (lanes<16) / {8..15,24..31} (lanes>=16)
__device__ inline v16h frag_b(const h16* W, int ldw, int n0, int k0) {
  int lane = threadIdx.x & 31;
  int lh = lane & 15, hb = (lane >> 4) << 3;
  const h16* r = W + (long)(n0 + lh) * ldw + k0 + hb;
  v16h f;
  #pragma unroll
  for (int i = 0; i < 8; ++i) { f[i] = r[i]; f[8 + i] = r[16 + i]; }
  return f;
}

// ---------------- utility kernels ----------------
__global__ void k_f32_to_f16(const float* __restrict__ s, h16* __restrict__ d, int n) {
  int i = blockIdx.x * blockDim.x + threadIdx.x;
  if (i < n) d[i] = (h16)s[i];
}
__global__ void k_transpose_f32f16(const float* __restrict__ s, h16* __restrict__ d,
                                   int R, int C, long ss_, long ds_) {
  long i = blockIdx.x * (long)blockDim.x + threadIdx.x;
  const float* sp = s + (long)blockIdx.y * ss_;
  h16* dp = d + (long)blockIdx.y * ds_;
  if (i < (long)R * C) { int r = (int)(i / C), c = (int)(i % C); dp[(long)c * R + r] = (h16)sp[i]; }
}

// exact top-512-of-4096 by rank (stable ties like jax.lax.top_k), per batch row
__global__ void k_topk(const float* __restrict__ x, float* __restrict__ vals, int* __restrict__ idx) {
  __shared__ float xl[CF];
  int b = blockIdx.x;
  const float* xr = x + (long)b * CF;
  for (int i = threadIdx.x; i < CF; i += blockDim.x) {
    float t = xr[i]; t = t < 0.f ? 0.f : t; xl[i] = log1pf(t);
  }
  __syncthreads();
  for (int t = 0; t < CF; t += blockDim.x) {
    int p = t + threadIdx.x;
    float vp = xl[p];
    int rank = 0;
    for (int q = 0; q < CF; ++q) {
      float vq = xl[q];
      rank += (vq > vp) || (vq == vp && q < p);
    }
    if (rank < CTK) { vals[b * CTK + rank] = vp; idx[b * CTK + rank] = p; }
  }
}

// encoder: roles/filler/outer product -> outer16 [B*CTK, 256] f16
__global__ void k_encode(const float* __restrict__ x, const float* __restrict__ mz,
                         const float* __restrict__ vals, const int* __restrict__ idx,
                         const float* __restrict__ emb, const float* __restrict__ wfil,
                         const float* __restrict__ cw, const float* __restrict__ cb,
                         h16* __restrict__ outer) {
  int gid = blockIdx.x * blockDim.x + threadIdx.x;  // b*CTK + k
  if (gid >= B_ * CTK) return;
  int b = gid >> 9;
  int p = idx[gid];
  float val = vals[gid];
  const float* xr = x + (long)b * CF;
  float xw[5];
  #pragma unroll
  for (int t = 0; t < 5; ++t) {
    int q = p + t - 2;
    q = q < 0 ? -q : (q > CF - 1 ? 2 * (CF - 1) - q : q);   // reflect pad
    float xx = xr[q]; xx = xx < 0.f ? 0.f : xx;
    xw[t] = log1pf(xx);
  }
  float roles[16], fill[16];
  float pos = mz[p];
  const float lf = -9.210340371976184f / 7.f;  // -ln(10000)/(half-1)
  #pragma unroll
  for (int j = 0; j < 8; ++j) {
    float e = pos * expf((float)j * lf);
    roles[j]     = sinf(e) + emb[p * 16 + j];
    roles[8 + j] = cosf(e) + emb[p * 16 + 8 + j];
  }
  #pragma unroll
  for (int o = 0; o < 16; ++o) {
    float ff = cb[o];
    #pragma unroll
    for (int t = 0; t < 5; ++t) ff += xw[t] * cw[o * 5 + t];
    ff = gelu_f(ff);
    fill[o] = gelu_f(val * wfil[o] + ff);
  }
  h16* orow = outer + (long)gid * DM_;
  #pragma unroll
  for (int d = 0; d < 16; ++d)
    #pragma unroll
    for (int e = 0; e < 16; ++e)
      orow[d * 16 + e] = (h16)(roles[d] * fill[e]);
}

// generic batched WMMA GEMM: out = act(alpha * X@W^T + bias). batch z: off = (z/bi)*so + (z%bi)*si
__global__ __launch_bounds__(256) void k_gemm(
    const h16* __restrict__ X, int lda, long sxo, long sxi,
    const h16* __restrict__ W, int ldw, long swo, long swi,
    const float* __restrict__ bias,
    float* __restrict__ out32, h16* __restrict__ out16, int ldc, long soo, long soi,
    int M, int N, int K, int bi, float alpha, int act) {
  int wave = threadIdx.x >> 5, lane = threadIdx.x & 31;
  int tm_n = M >> 4, tn_n = N >> 4;
  int tile = blockIdx.x * 8 + wave;
  if (tile >= tm_n * tn_n) return;
  int tm = tile % tm_n, tn = tile / tm_n;
  int z = blockIdx.y, zo = z / bi, zi = z % bi;
  const h16* Xp = X + (long)zo * sxo + (long)zi * sxi;
  const h16* Wp = W + (long)zo * swo + (long)zi * swi;
  long oof = (long)zo * soo + (long)zi * soi;
  v8f acc = {};
  for (int k0 = 0; k0 < K; k0 += 32) {
    v16h a = frag_a(Xp, lda, tm << 4, k0);
    v16h b = frag_b(Wp, ldw, tn << 4, k0);
    acc = wmma16(a, b, acc);
  }
  int lh = lane & 15, mo = (lane >> 4) << 3;
  int n = (tn << 4) + lh;
  float bs = bias ? bias[n] : 0.f;
  #pragma unroll
  for (int v = 0; v < 8; ++v) {
    int m = (tm << 4) + mo + v;
    float r = acc[v] * alpha + bs;
    if (act == 1) r = gelu_f(r);
    long off = oof + (long)m * ldc + n;
    if (out32) out32[off] = r;
    if (out16) out16[off] = (h16)r;
  }
}

// row LayerNorm over 256 channels; optional residual; one wave per row
__global__ __launch_bounds__(256) void k_ln256(
    const float* __restrict__ a, const float* __restrict__ res,
    const float* __restrict__ g, const float* __restrict__ bb,
    float* __restrict__ out32, h16* __restrict__ out16, int rows) {
  int wave = threadIdx.x >> 5, lane = threadIdx.x & 31;
  int row = blockIdx.x * 8 + wave;
  if (row >= rows) return;
  long base = (long)row * DM_;
  float v[8];
  float s = 0.f;
  #pragma unroll
  for (int i = 0; i < 8; ++i) {
    int c = lane + (i << 5);
    float xx = a[base + c];
    if (res) xx += res[base + c];
    v[i] = xx; s += xx;
  }
  s = wsum(s);
  float mean = s * (1.f / 256.f);
  float vs = 0.f;
  #pragma unroll
  for (int i = 0; i < 8; ++i) { float d2 = v[i] - mean; vs += d2 * d2; }
  vs = wsum(vs);
  float inv = rsqrtf(vs * (1.f / 256.f) + 1e-5f);
  #pragma unroll
  for (int i = 0; i < 8; ++i) {
    int c = lane + (i << 5);
    float o = (v[i] - mean) * inv * g[c] + bb[c];
    out32[base + c] = o;
    if (out16) out16[base + c] = (h16)o;
  }
}

// fused sparse second-order attention. grid = B*H*8 blocks, 256 thr (8 waves, 8 queries each)
__global__ __launch_bounds__(256) void k_attn(
    const h16* __restrict__ q16, const h16* __restrict__ k16, const h16* __restrict__ v16t,
    const h16* __restrict__ qp16, const h16* __restrict__ kp16,
    const h16* __restrict__ w1f, const float* __restrict__ b1,
    const float* __restrict__ w2, const float* __restrict__ b2,
    h16* __restrict__ ao16) {
  __shared__ __align__(16) h16 sk[CTK * DH];   // 32 KB
  __shared__ float ss[8][CTK];                 // 16 KB
  int blk = blockIdx.x;
  int qb = blk & 7, bh = blk >> 3;
  int h = bh & 7, b = bh >> 3;
  int tid = threadIdx.x;
  const h16* kpb0 = kp16 + (long)bh * (CTK * DH);
  const h16* vbase = v16t + ((long)(b * CTK)) * DM_ + h * DH;
  // warm L2/L0 for the scattered kp / v reads later (global_prefetch_b8)
  __builtin_prefetch(kpb0 + tid * 64, 0, 3);
  __builtin_prefetch(v16t + ((long)(b * CTK + (tid << 1))) * DM_ + h * DH, 0, 3);
  // stage k tile via CDNA5 async global->LDS b128 copies (tracked by ASYNCcnt,
  // no VGPR round-trip). Low 32 bits of a generic LDS pointer == raw LDS offset.
  for (int i = tid * 8; i < CTK * DH; i += 256 * 8) {
    int j = i >> 5, d = i & 31;
    const h16* gp = k16 + ((long)(b * CTK + j)) * DM_ + h * DH + d;
    unsigned lo = (unsigned)(unsigned long long)&sk[i];
    asm volatile("global_load_async_to_lds_b128 %0, %1, off"
                 :: "v"(lo), "v"(gp) : "memory");
  }
  asm volatile("s_wait_asynccnt 0x0" ::: "memory");
  __syncthreads();
  int wave = tid >> 5, lane = tid & 31;
  int lh = lane & 15, hi = lane >> 4;
  int hb = hi << 3;
  // wqk[d,e] = w1[e, 64+d]  -> B fragments (e-tiles 0..15 and 16..31)
  v16h bfA = frag_b(w1f + 64, 96, 0, 0);
  v16h bfB = frag_b(w1f + 64, 96, 16, 0);
  float b1a = b1[lh], b1b = b1[lh + 16];
  float w2a = w2[lh], w2b = w2[lh + 16];
  float b2v = b2[0];
  float* s = ss[wave];
  const float scl = 0.17677669529663687f;  // 1/sqrt(32)
  for (int qi = 0; qi < 8; ++qi) {
    int iq = qb * 64 + wave * 8 + qi;
    long qrow = ((long)(b * CTK + iq)) * DM_ + h * DH;
    h16 qf[16];
    #pragma unroll
    for (int t = 0; t < 8; ++t) { qf[t] = q16[qrow + hb + t]; qf[8 + t] = q16[qrow + 16 + hb + t]; }
    long qpo = (long)bh * (CTK * DH) + (long)iq * DH;
    float qpa = (float)qp16[qpo + lh] + b1a;
    float qpb = (float)qp16[qpo + lh + 16] + b1b;
    for (int jt = 0; jt < 32; ++jt) {
      int j0 = jt << 4;
      const h16* kr = sk + (j0 + lh) * DH;
      v16h a;
      #pragma unroll
      for (int t = 0; t < 8; ++t) { a[t] = kr[hb + t] * qf[t]; a[8 + t] = kr[16 + hb + t] * qf[8 + t]; }
      // kp[j,e] folded into WMMA accumulator (C layout: elem v -> j=j0+hb+v, n=lh)
      v8f c0, c1;
      const h16* kpt = kpb0 + (long)(j0 + hb) * DH;
      #pragma unroll
      for (int v = 0; v < 8; ++v) { c0[v] = (float)kpt[v * DH + lh]; c1[v] = (float)kpt[v * DH + lh + 16]; }
      v8f d0 = wmma16(a, bfA, c0);
      v8f d1 = wmma16(a, bfB, c1);
      float part[8];
      #pragma unroll
      for (int v = 0; v < 8; ++v)
        part[v] = gelu_f(d0[v] + qpa) * w2a + gelu_f(d1[v] + qpb) * w2b;
      #pragma unroll
      for (int v = 0; v < 8; ++v) {   // reduce over the 16 lanes sharing the same j rows
        float pv = part[v];
        pv += __shfl_xor(pv, 1, 32);
        pv += __shfl_xor(pv, 2, 32);
        pv += __shfl_xor(pv, 4, 32);
        pv += __shfl_xor(pv, 8, 32);
        part[v] = pv;
      }
      if (lh == 0) {
        #pragma unroll
        for (int v = 0; v < 8; ++v) s[j0 + hb + v] = (part[v] + b2v) * scl;
      }
    }
    // exact 64th-largest threshold via bitwise search on order-preserving uint map
    float fv[16]; unsigned uv[16];
    #pragma unroll
    for (int t = 0; t < 16; ++t) {
      float xx = s[lane + (t << 5)];
      fv[t] = xx;
      unsigned u = __float_as_uint(xx);
      uv[t] = (u >> 31) ? ~u : (u | 0x80000000u);
    }
    unsigned thr = 0;
    for (int bit = 31; bit >= 0; --bit) {
      unsigned cand = thr | (1u << bit);
      int cnt = 0;
      #pragma unroll
      for (int t = 0; t < 16; ++t) cnt += (uv[t] >= cand);
      cnt = wsumi(cnt);
      if (cnt >= 64) thr = cand;
    }
    float mx = -3.4e38f;
    #pragma unroll
    for (int t = 0; t < 16; ++t) mx = fmaxf(mx, fv[t]);
    mx = wmaxf(mx);
    float sum = 0.f;
    #pragma unroll
    for (int t = 0; t < 16; ++t) {
      float e = (uv[t] >= thr) ? expf(fv[t] - mx) : 0.f;
      sum += e;
      s[lane + (t << 5)] = e;
    }
    sum = wsum(sum);
    float inv = 1.f / sum;
    const h16* vb = vbase + lane;
    float acc = 0.f;
    for (int j = 0; j < CTK; ++j) acc += s[j] * (float)vb[(long)j * DM_];
    ao16[qrow + lane] = (h16)(acc * inv);
  }
}

// row softmax (rows of length n), also write f16 mirror
__global__ __launch_bounds__(256) void k_softmax_rows(float* __restrict__ a, h16* __restrict__ a16,
                                                      int rows, int n) {
  int wave = threadIdx.x >> 5, lane = threadIdx.x & 31;
  int row = blockIdx.x * 8 + wave;
  if (row >= rows) return;
  float* r = a + (long)row * n;
  float m = -3.4e38f;
  for (int i = lane; i < n; i += 32) m = fmaxf(m, r[i]);
  m = wmaxf(m);
  float sum = 0.f;
  for (int i = lane; i < n; i += 32) { float e = expf(r[i] - m); r[i] = e; sum += e; }
  sum = wsum(sum);
  float inv = 1.f / sum;
  for (int i = lane; i < n; i += 32) { float e = r[i] * inv; r[i] = e; a16[(long)row * n + i] = (h16)e; }
}

__global__ void k_gru(const float* __restrict__ gi, const float* __restrict__ gh,
                      const float* __restrict__ slots, float* __restrict__ upd) {
  int gid = blockIdx.x * blockDim.x + threadIdx.x;
  if (gid >= B_ * NSL * DM_) return;
  int c = gid & 255;
  int sb = gid >> 8;
  int sl = sb & 63;
  const float* gir = gi + (long)sb * 768;
  const float* ghr = gh + (long)sl * 768;
  float r = sigm(gir[c] + ghr[c]);
  float z = sigm(gir[256 + c] + ghr[256 + c]);
  float nn = tanhf(gir[512 + c] + r * ghr[512 + c]);
  float hp = slots[sl * 256 + c];
  upd[gid] = (1.f - z) * nn + z * hp;
}

__global__ void k_mem(const float* __restrict__ upd, const float* __restrict__ g,
                      const float* __restrict__ bb, float* __restrict__ mem) {
  int b = blockIdx.x, c = threadIdx.x;  // 256 threads
  float m = 0.f;
  for (int sl = 0; sl < NSL; ++sl) m += upd[((long)(b * NSL + sl)) * DM_ + c];
  m *= (1.f / 64.f);
  __shared__ float red[256];
  red[c] = m;
  __syncthreads();
  for (int st = 128; st; st >>= 1) { if (c < st) red[c] += red[c + st]; __syncthreads(); }
  float mean = red[0] * (1.f / 256.f);
  __syncthreads();
  float d2 = m - mean;
  red[c] = d2 * d2;
  __syncthreads();
  for (int st = 128; st; st >>= 1) { if (c < st) red[c] += red[c + st]; __syncthreads(); }
  float var = red[0] * (1.f / 256.f);
  mem[b * DM_ + c] = (m - mean) * rsqrtf(var + 1e-5f) * g[c] + bb[c];
}

__global__ void k_readout(const float* __restrict__ bact, const float* __restrict__ tqp,
                          const float* __restrict__ mem, const float* __restrict__ hw,
                          const float* __restrict__ hbv, float* __restrict__ out) {
  int b = blockIdx.x, tid = threadIdx.x;  // 256 threads
  __shared__ float aw[CTK];
  __shared__ float tq[DM_];
  __shared__ float pooled[DM_];
  __shared__ float sms;
  tq[tid] = tqp[tid];
  __syncthreads();
  for (int j = tid; j < CTK; j += 256) {
    const float* br = bact + ((long)(b * CTK + j)) * DM_;
    float dot = 0.f;
    for (int d = 0; d < DM_; ++d) dot += tq[d] * br[d];
    aw[j] = dot * (1.f / 16.f);
  }
  __syncthreads();
  if (tid == 0) {
    float m = -3.4e38f;
    for (int j = 0; j < CTK; ++j) m = fmaxf(m, aw[j]);
    float s2 = 0.f;
    for (int j = 0; j < CTK; ++j) { float e = expf(aw[j] - m); aw[j] = e; s2 += e; }
    sms = s2;
  }
  __syncthreads();
  float inv = 1.f / sms;
  float acc = 0.f;
  for (int j = 0; j < CTK; ++j) acc += aw[j] * bact[((long)(b * CTK + j)) * DM_ + tid];
  pooled[tid] = acc * inv + mem[b * DM_ + tid];
  __syncthreads();
  if (tid < 10) {
    float o = hbv[tid];
    for (int d = 0; d < DM_; ++d) o += pooled[d] * hw[tid * DM_ + d];
    out[b * 10 + tid] = o;
  }
}

// ---------------- host ----------------
extern "C" void kernel_launch(void* const* d_in, const int* in_sizes, int n_in,
                              void* d_out, int out_size, void* d_ws, size_t ws_size,
                              hipStream_t stream) {
  (void)out_size; (void)ws_size;
  struct Layer { const float *wq,*bq,*wk,*bk,*wv,*bv,*wo,*bo,*w1,*b1,*w2,*b2,
                             *n1g,*n1b,*n2g,*n2b,*wf1,*bf1,*wf2,*bf2; };
  const float *x,*mz,*emb,*wfil,*cw,*cb,*wout,*eg,*eb,*slots,*gwih,*gwhh,*gbih,*gbhh,
              *mg,*mb,*tq,*hw,*hb;
  Layer L[3];
  auto F = [&](int i) { return (const float*)d_in[i]; };
  if (n_in >= 1 && in_sizes[0] == B_ * CF) {
    // insertion order
    x=F(0); mz=F(1); emb=F(2); wfil=F(3); cw=F(4); cb=F(5); wout=F(6); eg=F(7); eb=F(8);
    int o = 9;
    for (int l = 0; l < 3; l++, o += 20) {
      L[l].wq=F(o+0);  L[l].bq=F(o+1);  L[l].wk=F(o+2);  L[l].bk=F(o+3);
      L[l].wv=F(o+4);  L[l].bv=F(o+5);  L[l].wo=F(o+6);  L[l].bo=F(o+7);
      L[l].w1=F(o+8);  L[l].b1=F(o+9);  L[l].w2=F(o+10); L[l].b2=F(o+11);
      L[l].n1g=F(o+12);L[l].n1b=F(o+13);L[l].n2g=F(o+14);L[l].n2b=F(o+15);
      L[l].wf1=F(o+16);L[l].bf1=F(o+17);L[l].wf2=F(o+18);L[l].bf2=F(o+19);
    }
    slots=F(o+0); gwih=F(o+1); gwhh=F(o+2); gbih=F(o+3); gbhh=F(o+4);
    mg=F(o+5); mb=F(o+6); tq=F(o+7); hw=F(o+8); hb=F(o+9);
  } else {
    // jax tree-flatten (sorted keys) leaf order
    mz=F(0); cb=F(1); cw=F(2); emb=F(3); eb=F(4); eg=F(5);
    gbhh=F(6); gbih=F(7); gwhh=F(8); gwih=F(9); hb=F(10); hw=F(11);
    int o = 12;
    for (int l = 0; l < 3; l++, o += 20) {
      L[l].b1=F(o+0);  L[l].b2=F(o+1);  L[l].bf1=F(o+2); L[l].bf2=F(o+3);
      L[l].bk=F(o+4);  L[l].bo=F(o+5);  L[l].bq=F(o+6);  L[l].bv=F(o+7);
      L[l].n1b=F(o+8); L[l].n1g=F(o+9); L[l].n2b=F(o+10);L[l].n2g=F(o+11);
      L[l].w1=F(o+12); L[l].w2=F(o+13); L[l].wf1=F(o+14);L[l].wf2=F(o+15);
      L[l].wk=F(o+16); L[l].wo=F(o+17); L[l].wq=F(o+18); L[l].wv=F(o+19);
    }
    mb=F(o+0); mg=F(o+1); slots=F(o+2); tq=F(o+3); wfil=F(o+4); wout=F(o+5); x=F(o+6);
  }

  // ---- workspace arena ----
  char* wsb = (char*)d_ws;
  size_t cur = 0;
  auto alloc = [&](size_t bytes) -> void* {
    void* r = wsb + cur; cur = (cur + bytes + 255) & ~(size_t)255; return r;
  };
  const size_t TOK = (size_t)B_ * CTK;          // 4096 tokens
  float* vals = (float*)alloc(TOK * 4);
  int*   idx  = (int*)  alloc(TOK * 4);
  h16* outer16 = (h16*)alloc(TOK * DM_ * 2);
  float* tmp1  = (float*)alloc(TOK * DM_ * 4);
  float* bbuf  = (float*)alloc(TOK * DM_ * 4);
  h16* bbuf16  = (h16*)alloc(TOK * DM_ * 2);
  h16* q16  = (h16*)alloc(TOK * DM_ * 2);
  h16* k16  = (h16*)alloc(TOK * DM_ * 2);
  h16* v16  = (h16*)alloc(TOK * DM_ * 2);
  h16* qp16 = (h16*)alloc(TOK * DM_ * 2);       // [B*H,512,32]
  h16* kp16 = (h16*)alloc(TOK * DM_ * 2);
  h16* ao16 = (h16*)alloc(TOK * DM_ * 2);
  h16* h116 = (h16*)alloc(TOK * 1024 * 2);
  h16* wout16 = (h16*)alloc(65536 * 2);
  h16 *wq16[3], *wk16[3], *wv16[3], *wo16[3], *w116[3], *wf116[3], *wf216[3];
  for (int l = 0; l < 3; l++) {
    wq16[l]  = (h16*)alloc(65536 * 2);
    wk16[l]  = (h16*)alloc(65536 * 2);
    wv16[l]  = (h16*)alloc(65536 * 2);
    wo16[l]  = (h16*)alloc(65536 * 2);
    w116[l]  = (h16*)alloc(3072 * 2);
    wf116[l] = (h16*)alloc(262144 * 2);
    wf216[l] = (h16*)alloc(262144 * 2);
  }
  h16* slots16 = (h16*)alloc(NSL * DM_ * 2);
  h16* gwih16  = (h16*)alloc(768 * 256 * 2);
  h16* gwhh16  = (h16*)alloc(768 * 256 * 2);
  float* sc  = (float*)alloc((size_t)B_ * NSL * CTK * 4);
  h16* A16   = (h16*)alloc((size_t)B_ * NSL * CTK * 2);
  h16* bT16  = (h16*)alloc((size_t)B_ * DM_ * CTK * 2);
  h16* ctx16 = (h16*)alloc((size_t)B_ * NSL * DM_ * 2);
  float* gi  = (float*)alloc((size_t)B_ * NSL * 768 * 4);
  float* gh0 = (float*)alloc((size_t)NSL * 768 * 4);
  float* upd = (float*)alloc((size_t)B_ * NSL * DM_ * 4);
  float* memv = (float*)alloc((size_t)B_ * DM_ * 4);

  auto cvt = [&](const float* s, h16* d, int n) {
    k_f32_to_f16<<<(n + 255) / 256, 256, 0, stream>>>(s, d, n);
  };
  auto gemm = [&](const h16* X, int lda, long sxo, long sxi,
                  const h16* W, int ldw, long swo, long swi,
                  const float* bias, float* o32, h16* o16, int ldc, long soo, long soi,
                  int M, int N, int K, int batch, int bi, float alpha, int act) {
    int tiles = (M / 16) * (N / 16);
    k_gemm<<<dim3((tiles + 7) / 8, batch), 256, 0, stream>>>(
        X, lda, sxo, sxi, W, ldw, swo, swi, bias, o32, o16, ldc, soo, soi,
        M, N, K, bi, alpha, act);
  };

  // ---- pipeline ----
  k_topk<<<B_, 512, 0, stream>>>(x, vals, idx);
  k_encode<<<(B_ * CTK) / 256, 256, 0, stream>>>(x, mz, vals, idx, emb, wfil, cw, cb, outer16);
  cvt(wout, wout16, 65536);
  cvt(slots, slots16, NSL * DM_);
  cvt(gwih, gwih16, 768 * 256);
  cvt(gwhh, gwhh16, 768 * 256);
  for (int l = 0; l < 3; l++) {
    cvt(L[l].wq, wq16[l], 65536);  cvt(L[l].wk, wk16[l], 65536);
    cvt(L[l].wv, wv16[l], 65536);  cvt(L[l].wo, wo16[l], 65536);
    cvt(L[l].w1, w116[l], 3072);
    cvt(L[l].wf1, wf116[l], 262144); cvt(L[l].wf2, wf216[l], 262144);
  }
  // encoder projection + LN
  gemm(outer16, 256, 0, 0, wout16, 256, 0, 0, nullptr,
       tmp1, nullptr, 256, 0, 0, 4096, 256, 256, 1, 1, 1.f, 0);
  k_ln256<<<512, 256, 0, stream>>>(tmp1, nullptr, eg, eb, bbuf, bbuf16, 4096);

  for (int l = 0; l < 3; l++) {
    gemm(bbuf16, 256, 0, 0, wq16[l], 256, 0, 0, L[l].bq,
         nullptr, q16, 256, 0, 0, 4096, 256, 256, 1, 1, 1.f, 0);
    gemm(bbuf16, 256, 0, 0, wk16[l], 256, 0, 0, L[l].bk,
         nullptr, k16, 256, 0, 0, 4096, 256, 256, 1, 1, 1.f, 0);
    gemm(bbuf16, 256, 0, 0, wv16[l], 256, 0, 0, L[l].bv,
         nullptr, v16, 256, 0, 0, 4096, 256, 256, 1, 1, 1.f, 0);
    // qp/kp: per-(b,h) batched (z = b*8+h): X off = b*512*256 + h*32 ; out [bh,512,32]
    gemm(q16, 256, 131072, 32, w116[l] + 0, 96, 0, 0, nullptr,
         nullptr, qp16, 32, 131072, 16384, 512, 32, 32, 64, 8, 1.f, 0);
    gemm(k16, 256, 131072, 32, w116[l] + 32, 96, 0, 0, nullptr,
         nullptr, kp16, 32, 131072, 16384, 512, 32, 32, 64, 8, 1.f, 0);
    k_attn<<<512, 256, 0, stream>>>(q16, k16, v16, qp16, kp16,
                                    w116[l], L[l].b1, L[l].w2, L[l].b2, ao16);
    gemm(ao16, 256, 0, 0, wo16[l], 256, 0, 0, L[l].bo,
         tmp1, nullptr, 256, 0, 0, 4096, 256, 256, 1, 1, 1.f, 0);
    k_ln256<<<512, 256, 0, stream>>>(tmp1, bbuf, L[l].n1g, L[l].n1b, bbuf, bbuf16, 4096);
    gemm(bbuf16, 256, 0, 0, wf116[l], 256, 0, 0, L[l].bf1,
         nullptr, h116, 1024, 0, 0, 4096, 1024, 256, 1, 1, 1.f, 1);
    gemm(h116, 1024, 0, 0, wf216[l], 1024, 0, 0, L[l].bf2,
         tmp1, nullptr, 256, 0, 0, 4096, 256, 1024, 1, 1, 1.f, 0);
    k_ln256<<<512, 256, 0, stream>>>(tmp1, bbuf, L[l].n2g, L[l].n2b, bbuf, bbuf16, 4096);
  }

  // gated relational memory
  gemm(slots16, 256, 0, 0, bbuf16, 256, 131072, 0, nullptr,
       sc, nullptr, 512, 32768, 0, 64, 512, 256, 8, 1, 0.0625f, 0);
  k_softmax_rows<<<64, 256, 0, stream>>>(sc, A16, 512, 512);
  k_transpose_f32f16<<<dim3(512, 8), 256, 0, stream>>>(bbuf, bT16, 512, 256, 131072L, 131072L);
  gemm(A16, 512, 32768, 0, bT16, 512, 131072, 0, nullptr,
       nullptr, ctx16, 256, 16384, 0, 64, 256, 512, 8, 1, 1.f, 0);
  gemm(ctx16, 256, 0, 0, gwih16, 256, 0, 0, gbih,
       gi, nullptr, 768, 0, 0, 512, 768, 256, 1, 1, 1.f, 0);
  gemm(slots16, 256, 0, 0, gwhh16, 256, 0, 0, gbhh,
       gh0, nullptr, 768, 0, 0, 64, 768, 256, 1, 1, 1.f, 0);
  k_gru<<<(B_ * NSL * DM_ + 255) / 256, 256, 0, stream>>>(gi, gh0, slots, upd);
  k_mem<<<B_, 256, 0, stream>>>(upd, mg, mb, memv);
  k_readout<<<B_, 256, 0, stream>>>(bbuf, tq, memv, hw, hb, (float*)d_out);
}
